// CategorySpecificLinear_5815385718920
// MI455X (gfx1250) — matrix operand order repaired and checked by reading, not compiled, not run on hardware
//
#include <hip/hip_runtime.h>

// Problem constants (match reference)
#define NUM_CATEGORIES 32
#define INPUT_DIM 1024
#define HIDDEN_DIM 2048
#define BATCH 64
#define SEQ 256
#define NK (INPUT_DIM / 32)        // 32 K-steps of 32

// Vector types
typedef _Float16 h2   __attribute__((ext_vector_type(2)));
typedef _Float16 h4   __attribute__((ext_vector_type(4)));
typedef _Float16 h8   __attribute__((ext_vector_type(8)));
typedef _Float16 v16h __attribute__((ext_vector_type(16)));
typedef float    v8f  __attribute__((ext_vector_type(8)));
typedef float    f4   __attribute__((ext_vector_type(4)));

// 32 halves of K per row + 8 pad halves -> 80B row pitch: 16B-aligned fragment
// reads, staggered banks.
#define LDS_STRIDE 40
#define TILE_HALVES (128 * LDS_STRIDE)

__global__ __launch_bounds__(256) void cat_linear_f16split_wmma(
    const float* __restrict__ x,        // [B, T, INPUT_DIM]
    const int*   __restrict__ cat_ids,  // [B]
    const float* __restrict__ W,        // [NUM_CATEGORIES, INPUT_DIM, HIDDEN_DIM]
    const float* __restrict__ bias,     // [NUM_CATEGORIES, HIDDEN_DIM]
    float*       __restrict__ out)      // [B, T, HIDDEN_DIM]
{
    // Double-buffered LDS: A (x tile) [t][k] row-major; B (W tile) transposed [h][k]
    __shared__ __align__(16) _Float16 Ahi[2][TILE_HALVES];
    __shared__ __align__(16) _Float16 Alo[2][TILE_HALVES];
    __shared__ __align__(16) _Float16 Bhi[2][TILE_HALVES];
    __shared__ __align__(16) _Float16 Blo[2][TILE_HALVES];

    const int tid   = threadIdx.x;
    const int lane  = tid & 31;
    const int wave  = tid >> 5;       // 8 wave32s
    const int l16   = lane & 15;
    const int lhi   = lane >> 4;      // lane half (0/1)
    const int waveM = wave & 3;       // 4 waves along M (t): 4*32 = 128
    const int waveN = wave >> 2;      // 2 waves along N (h): 2*64 = 128

    const int hbase = blockIdx.x * 128;
    const int tbase = blockIdx.y * 128;
    const int b     = blockIdx.z;
    const int cat   = cat_ids[b];

    const float* xb = x + ((size_t)b * SEQ + tbase) * INPUT_DIM;
    const float* Wc = W + (size_t)cat * INPUT_DIM * HIDDEN_DIM + hbase;

    // ---- Per-thread staging coordinates (K-invariant) ----
    // A: 128x32 f32 = 1024 float4, 4 per thread; 8 float4 per row
    const int arow = tid >> 3;        // + 32*r
    const int ac4  = tid & 7;
    // B: 32x128 f32; thread owns 2 (k,k+1) row-pairs x one h4 group
    const int bkp  = tid >> 5;        // k-pair index, + 8*r
    const int bh4  = tid & 31;        // h4 group (4 consecutive h)

    v8f acc[2][4];
#pragma unroll
    for (int mt = 0; mt < 2; ++mt)
#pragma unroll
        for (int nt = 0; nt < 4; ++nt)
#pragma unroll
            for (int e = 0; e < 8; ++e) acc[mt][nt][e] = 0.0f;

    // Prefetch registers for the software pipeline
    f4 aR[4];
    f4 bR[2][2];

    auto global_fetch = [&](int k0) {
#pragma unroll
        for (int r = 0; r < 4; ++r)
            aR[r] = *(const f4*)(xb + (size_t)(arow + 32 * r) * INPUT_DIM + k0 + ac4 * 4);
#pragma unroll
        for (int r = 0; r < 2; ++r) {
            const float* p = Wc + (size_t)(k0 + 2 * (bkp + 8 * r)) * HIDDEN_DIM + bh4 * 4;
            bR[r][0] = *(const f4*)(p);
            bR[r][1] = *(const f4*)(p + HIDDEN_DIM);
        }
    };

    auto lds_store = [&](int buf) {
        // A: hi/lo f16, 8-byte packed stores
#pragma unroll
        for (int r = 0; r < 4; ++r) {
            h4 hi, lo;
#pragma unroll
            for (int e = 0; e < 4; ++e) {
                float v = aR[r][e];
                _Float16 hh = (_Float16)v;
                hi[e] = hh;
                lo[e] = (_Float16)(v - (float)hh);
            }
            int off = (arow + 32 * r) * LDS_STRIDE + ac4 * 4;
            *(h4*)&Ahi[buf][off] = hi;
            *(h4*)&Alo[buf][off] = lo;
        }
        // B: transpose [k][h] -> [h][k]; pack (k, k+1) pairs into b32 stores
#pragma unroll
        for (int r = 0; r < 2; ++r) {
            int kcol = 2 * (bkp + 8 * r);
#pragma unroll
            for (int e = 0; e < 4; ++e) {
                float v0 = bR[r][0][e];
                float v1 = bR[r][1][e];
                _Float16 h0 = (_Float16)v0;
                _Float16 h1 = (_Float16)v1;
                _Float16 l0 = (_Float16)(v0 - (float)h0);
                _Float16 l1 = (_Float16)(v1 - (float)h1);
                int off = (bh4 * 4 + e) * LDS_STRIDE + kcol;
                *(h2*)&Bhi[buf][off] = h2{h0, h1};
                *(h2*)&Blo[buf][off] = h2{l0, l1};
            }
        }
    };

    // ---- Fragment LDS offsets (K-invariant) ----
    int aOff[2], bOff[4];
#pragma unroll
    for (int mt = 0; mt < 2; ++mt)
        aOff[mt] = (waveM * 32 + mt * 16 + l16) * LDS_STRIDE + lhi * 16;
#pragma unroll
    for (int nt = 0; nt < 4; ++nt)
        bOff[nt] = (waveN * 64 + nt * 16 + l16) * LDS_STRIDE + lhi * 16;

    // ---- Pipelined main loop: 1 barrier / iteration ----
    global_fetch(0);
    lds_store(0);

    for (int kk = 0; kk < NK; ++kk) {
        const int cb = kk & 1;
        if (kk + 1 < NK) global_fetch((kk + 1) * 32);  // in flight during compute

        __syncthreads();  // buf[cb] ready; readers of buf[cb^1] (iter kk-1) done

        v16h aH[2], aL[2], bH[4], bL[4];
#pragma unroll
        for (int mt = 0; mt < 2; ++mt) {
            ((h8*)&aH[mt])[0] = *(const h8*)&Ahi[cb][aOff[mt]];
            ((h8*)&aH[mt])[1] = *(const h8*)&Ahi[cb][aOff[mt] + 8];
            ((h8*)&aL[mt])[0] = *(const h8*)&Alo[cb][aOff[mt]];
            ((h8*)&aL[mt])[1] = *(const h8*)&Alo[cb][aOff[mt] + 8];
        }
#pragma unroll
        for (int nt = 0; nt < 4; ++nt) {
            ((h8*)&bH[nt])[0] = *(const h8*)&Bhi[cb][bOff[nt]];
            ((h8*)&bH[nt])[1] = *(const h8*)&Bhi[cb][bOff[nt] + 8];
            ((h8*)&bL[nt])[0] = *(const h8*)&Blo[cb][bOff[nt]];
            ((h8*)&bL[nt])[1] = *(const h8*)&Blo[cb][bOff[nt] + 8];
        }

        // 3-term f32-emulation: hi*hi + lo*hi + hi*lo (fp32 accumulate)
#pragma unroll
        for (int mt = 0; mt < 2; ++mt)
#pragma unroll
            for (int nt = 0; nt < 4; ++nt) {
                acc[mt][nt] = __builtin_amdgcn_wmma_f32_16x16x32_f16(
                    false, aH[mt], false, bH[nt], (short)0, acc[mt][nt], false, false);
                acc[mt][nt] = __builtin_amdgcn_wmma_f32_16x16x32_f16(
                    false, aL[mt], false, bH[nt], (short)0, acc[mt][nt], false, false);
                acc[mt][nt] = __builtin_amdgcn_wmma_f32_16x16x32_f16(
                    false, aH[mt], false, bL[nt], (short)0, acc[mt][nt], false, false);
            }

        if (kk + 1 < NK) lds_store(cb ^ 1);  // overlaps other waves' compute
    }

    // ---- Epilogue: bias + store (C/D layout: VGPR r -> M = r + 8*lhi, N = l16) ----
#pragma unroll
    for (int nt = 0; nt < 4; ++nt) {
        int h = hbase + waveN * 64 + nt * 16 + l16;
        float bv = bias[(size_t)cat * HIDDEN_DIM + h];
#pragma unroll
        for (int mt = 0; mt < 2; ++mt) {
            int trow0 = tbase + waveM * 32 + mt * 16 + 8 * lhi;
#pragma unroll
            for (int r = 0; r < 8; ++r) {
                out[((size_t)b * SEQ + trow0 + r) * HIDDEN_DIM + h] = acc[mt][nt][r] + bv;
            }
        }
    }
}

extern "C" void kernel_launch(void* const* d_in, const int* in_sizes, int n_in,
                              void* d_out, int out_size, void* d_ws, size_t ws_size,
                              hipStream_t stream) {
    const float* x       = (const float*)d_in[0];
    const int*   cat_ids = (const int*)d_in[1];
    const float* W       = (const float*)d_in[2];
    const float* b       = (const float*)d_in[3];
    float*       out     = (float*)d_out;

    dim3 grid(HIDDEN_DIM / 128, SEQ / 128, BATCH);  // (16, 2, 64)
    dim3 block(256);                                 // 8 wave32s
    cat_linear_f16split_wmma<<<grid, block, 0, stream>>>(x, cat_ids, W, b, out);
}
